// LipikaTokenizer_54004918780626
// MI455X (gfx1250) — compile-verified, matrix operand order checked by reading, not run on hardware
//
#include <hip/hip_runtime.h>
#include <hip/hip_bf16.h>
#include <cmath>

// ---------------------------------------------------------------------------
// LipikaTokenizer (VQ-VAE audio codec) for MI455X / gfx1250.
// One implicit-GEMM kernel using v_wmma_f32_16x16x32_f16 with 2x2 register
// blocking (4 WMMA / K-step / wave) and async global->LDS staging of the
// weight tile overlapped with the im2col gather (ASYNCcnt-fenced).
//   - convs / transposed convs: implicit im2col (+fused ELU) in the B loader
//   - RVQ distances: plain GEMM  (E_f16 [256x64]) x (residual [64x800])
//   - STFT: DFT-matrix GEMMs (cos/sin) over framed signals
// Activations stay fp32 in HBM; weights / DFT / codebooks packed to f16.
// ---------------------------------------------------------------------------

typedef __attribute__((ext_vector_type(16))) _Float16 v16h;
typedef __attribute__((ext_vector_type(8)))  float    v8f;
typedef __attribute__((ext_vector_type(4)))  int      v4i;

#define TM 64      // macro-tile rows
#define TNB 128    // macro-tile cols
#define TK 32      // K step

#if defined(__has_builtin)
#if __has_builtin(__builtin_amdgcn_global_load_async_to_lds_b128) && \
    __has_builtin(__builtin_amdgcn_s_wait_asynccnt)
#define USE_ASYNC_LDS 1
#endif
#endif

struct ConvDesc {
  int mode;    // 0 = plain f32 [K][N] matrix, 1 = implicit im2col conv
  int Cin, Lin, ksz, stride, padL, dil, lhsDil, elu;
};

__device__ __forceinline__ float eluf(float x) { return x > 0.f ? x : __expf(x) - 1.f; }

// ---------------------------------------------------------------------------
// Unified WMMA GEMM:  C[b][m][n] (+)= act( A_f16[m][k] * B(b)[k][n] + bias[m] )
// A: f16, row-major [Ma][K]   (K multiple of 32; rows >= Ma read as 0)
// B: mode 0: fp32 [K][N] row-major (per batch, strideB)
//    mode 1: implicit im2col from fp32 [Cin][Lin] with stride/pad/dil/lhsDil/ELU
// Block: 256 threads = 8 waves; wave (wm,wn) in 2x4 grid computes 32x32 out.
// ---------------------------------------------------------------------------
__global__ __launch_bounds__(256)
void k_gemm(const _Float16* __restrict__ A, const float* __restrict__ Bsrc,
            const float* __restrict__ bias, float* __restrict__ C,
            int Ma, int Mact, int N, int K,
            long strideB, long strideC, ConvDesc cd, int act, int accum)
{
  __shared__ _Float16 sA[TM][40];    // row stride 80B keeps 16B alignment
  __shared__ _Float16 sB[TNB][40];   // B tile stored transposed: [n][k]

  const int tid = threadIdx.x;
  const int b   = blockIdx.z;
  const int m0  = blockIdx.y * TM;
  const int n0  = blockIdx.x * TNB;
  const float* Bb = Bsrc + (long)b * strideB;
  float*       Cb = C    + (long)b * strideC;

  const int lane = tid & 31;
  const int wid  = tid >> 5;
  const int wm   = wid >> 2;          // 2 waves along M (32 rows each)
  const int wn   = wid & 3;           // 4 waves along N (32 cols each)
  const int lm   = lane & 15;
  const int hf   = lane >> 4;         // K-half group owned by this lane

  v8f acc[2][2];
  #pragma unroll
  for (int s = 0; s < 2; ++s)
    #pragma unroll
    for (int t = 0; t < 2; ++t)
      acc[s][t] = v8f{0.f, 0.f, 0.f, 0.f, 0.f, 0.f, 0.f, 0.f};

  const int am = tid >> 2;            // 0..63  A row within tile
  const int ak = (tid & 3) * 8;       // 8 halves (16B) per thread
  const int bk = tid >> 3;            // 0..31  B row (k) within tile
  const int bn = (tid & 7) * 4;       // 4 cols per thread per 32-col chunk

  for (int kt = 0; kt < K; kt += TK) {
    __syncthreads();

    // ---- A tile: 64x32 f16 = 4KB; one 16B op per thread ----
    {
      int gm = m0 + am;
      if (gm < Ma) {
#ifdef USE_ASYNC_LDS
        // CDNA5 async DMA: global -> LDS without touching VGPR data path;
        // overlapped with the im2col B-gather below, fenced by ASYNCcnt.
        __builtin_amdgcn_global_load_async_to_lds_b128(
            (v4i*)(A + (long)gm * K + kt + ak),
            (__attribute__((address_space(3))) v4i*)&sA[am][ak], 0, 0);
#else
        const _Float16* p = A + (long)gm * K + kt + ak;
        #pragma unroll
        for (int i = 0; i < 8; ++i) sA[am][ak + i] = p[i];
#endif
      } else {
        #pragma unroll
        for (int i = 0; i < 8; ++i) sA[am][ak + i] = (_Float16)0.f;
      }
    }

    // ---- B tile: 32(k) x 128(n), gathered fp32 -> f16, stored [n][k] ----
    {
      int row = kt + bk;
      int ci = 0, kk = 0, inRange = 0;
      if (cd.mode == 1) {
        inRange = row < cd.Cin * cd.ksz;
        if (inRange) { ci = row / cd.ksz; kk = row % cd.ksz; }
      }
      #pragma unroll
      for (int c = 0; c < 4; ++c) {
        int nl = c * 32 + bn;          // col within macro-tile
        #pragma unroll
        for (int i = 0; i < 4; ++i) {
          int n = n0 + nl + i;
          float v = 0.f;
          if (n < N) {
            if (cd.mode == 0) {
              v = Bb[(long)row * N + n];
            } else if (inRange) {
              int p = n * cd.stride - cd.padL + kk * cd.dil;
              if (cd.lhsDil > 1) {               // transposed conv (lhs dilation)
                if (p >= 0) {
                  int q = p / cd.lhsDil;
                  if (q * cd.lhsDil == p && q < cd.Lin) v = Bb[(long)ci * cd.Lin + q];
                }
              } else if (p >= 0 && p < cd.Lin) {
                v = Bb[(long)ci * cd.Lin + p];
              }
              if (cd.elu) v = eluf(v);           // pre-conv ELU fused here
            }
          }
          sB[nl + i][bk] = (_Float16)v;
        }
      }
    }

#ifdef USE_ASYNC_LDS
    __builtin_amdgcn_s_wait_asynccnt(0);
#endif
    __syncthreads();

    // ---- fragments: two contiguous 8-half runs per lane, 2 per matrix ----
    v16h aF[2], bF[2];
    #pragma unroll
    for (int s = 0; s < 2; ++s) {
      const _Float16* pa = &sA[wm * 32 + s * 16 + lm][hf * 8];
      const _Float16* pb = &sB[wn * 32 + s * 16 + lm][hf * 8];
      #pragma unroll
      for (int i = 0; i < 8; ++i) {
        aF[s][i] = pa[i]; aF[s][8 + i] = pa[16 + i];
        bF[s][i] = pb[i]; bF[s][8 + i] = pb[16 + i];
      }
    }
    #pragma unroll
    for (int s = 0; s < 2; ++s)
      #pragma unroll
      for (int t = 0; t < 2; ++t)
        acc[s][t] = __builtin_amdgcn_wmma_f32_16x16x32_f16(
            false, aF[s], false, bF[t], (short)0, acc[s][t], false, false);
  }

  // ---- epilogue ----
  #pragma unroll
  for (int t = 0; t < 2; ++t) {
    int n = n0 + wn * 32 + t * 16 + lm;
    if (n < N) {
      #pragma unroll
      for (int s = 0; s < 2; ++s) {
        #pragma unroll
        for (int r = 0; r < 8; ++r) {
          int m = m0 + wm * 32 + s * 16 + hf * 8 + r;
          if (m < Mact) {
            float v = acc[s][t][r];
            if (bias) v += bias[m];
            if (act == 1) v = tanhf(v);
            long idx = (long)m * N + n;
            if (accum) Cb[idx] += v; else Cb[idx] = v;
          }
        }
      }
    }
  }
}

// ------------------------- small support kernels ---------------------------

// Pack conv weights to f16 GEMM-A layout [Ma][Kp], zero padded.
// flip=0: w is [Cout][Cin][k];  flip=1: w is [Cin][Cout][k], taps reversed (convT).
__global__ void k_pack_w(const float* __restrict__ w, _Float16* __restrict__ A,
                         int CoutAct, int Cin, int ksz, int Kp, int flip, int n)
{
  int i = blockIdx.x * blockDim.x + threadIdx.x;
  if (i >= n) return;
  int col = i % Kp, co = i / Kp;
  float v = 0.f;
  if (co < CoutAct && col < Cin * ksz) {
    int ci = col / ksz, kk = col % ksz;
    v = flip ? w[((long)ci * CoutAct + co) * ksz + (ksz - 1 - kk)]
             : w[((long)co * Cin + ci) * ksz + kk];
  }
  A[i] = (_Float16)v;
}

__global__ void k_enorm(const float* __restrict__ E, float* __restrict__ out, int n)
{
  int k = blockIdx.x * blockDim.x + threadIdx.x;
  if (k >= n) return;
  float s = 0.f;
  for (int d = 0; d < 64; ++d) { float e = E[(long)k * 64 + d]; s += e * e; }
  out[k] = s;
}

// Per (b,t): argmin_k(|E_k|^2 - 2 dot) ; update residual, zq sum, codes, sq-loss.
__global__ void k_vq(const float* __restrict__ dot, const float* __restrict__ enorm,
                     const float* __restrict__ E, float* __restrict__ r,
                     float* __restrict__ zq, float* __restrict__ codes,
                     float* __restrict__ accSq, int stage)
{
  int i = blockIdx.x * blockDim.x + threadIdx.x;
  if (i >= 4 * 800) return;
  int b = i / 800, t = i % 800;
  const float* db = dot + (long)b * 256 * 800 + t;
  int best = 0; float bestv = 3.4e38f;
  for (int k = 0; k < 256; ++k) {
    float v = enorm[k] - 2.f * db[(long)k * 800];
    if (v < bestv) { bestv = v; best = k; }
  }
  codes[((long)b * 800 + t) * 4 + stage] = (float)best;
  float s = 0.f;
  float* rb = r  + (long)b * 64 * 800 + t;
  float* zb = zq + (long)b * 64 * 800 + t;
  const float* e = E + (long)best * 64;
  for (int d = 0; d < 64; ++d) {
    float ev = e[d];
    float df = ev - rb[(long)d * 800];
    s += df * df;
    zb[(long)d * 800] += ev;
    rb[(long)d * 800] -= ev;
  }
  atomicAdd(accSq, s);
}

__global__ void k_zq_out(const float* __restrict__ zq, float* __restrict__ out)
{
  int i = blockIdx.x * blockDim.x + threadIdx.x;
  if (i >= 4 * 800 * 64) return;
  int b = i / (800 * 64), rem = i % (800 * 64), t = rem / 64, d = rem % 64;
  out[i] = zq[((long)b * 64 + d) * 800 + t];
}

// Frames (windowed, reflect-padded) stored transposed [n_fft][Ncols] for GEMM-B.
__global__ void k_frames(const float* __restrict__ x, float* __restrict__ FrT,
                         int n_fft, int hop, int F, int Ncols, int n)
{
  int i = blockIdx.x * blockDim.x + threadIdx.x;
  if (i >= n) return;
  int nn = i / Ncols, col = i % Ncols;
  int b = col / F, f = col % F;
  int j = f * hop + nn - n_fft / 2;
  if (j < 0) j = -j;
  if (j >= 80000) j = 2 * 80000 - 2 - j;
  float win = 0.5f - 0.5f * __cosf(6.2831853071795864f * (float)nn / (float)n_fft);
  FrT[(long)nn * Ncols + col] = x[(long)b * 80000 + j] * win;
}

__global__ void k_dftw(_Float16* __restrict__ Wc, _Float16* __restrict__ Ws,
                       int n_fft, int bins, int n)
{
  int i = blockIdx.x * blockDim.x + threadIdx.x;
  if (i >= n) return;
  int bin = i / n_fft, nn = i % n_fft;
  float c = 0.f, s = 0.f;
  if (bin < bins) {
    long ph = ((long)bin * nn) % n_fft;            // exact phase wrap
    float a = (float)ph * (6.2831853071795864f / (float)n_fft);
    __sincosf(a, &s, &c);
  }
  Wc[i] = (_Float16)c; Ws[i] = (_Float16)s;
}

__global__ void k_sloss(const float* __restrict__ xr, const float* __restrict__ xi,
                        const float* __restrict__ hr, const float* __restrict__ hi,
                        float* __restrict__ acc, float invCount, int n)
{
  int i = blockIdx.x * blockDim.x + threadIdx.x;
  if (i >= n) return;
  float a = sqrtf(xr[i] * xr[i] + xi[i] * xi[i]);
  float b = sqrtf(hr[i] * hr[i] + hi[i] * hi[i]);
  atomicAdd(acc, fabsf(a - b) * invCount);
}

__global__ void k_final(const float* __restrict__ accs, float* __restrict__ out)
{
  if (threadIdx.x == 0 && blockIdx.x == 0) {
    float vq = 1.25f * accs[0] / (4.f * 800.f * 64.f);   // (1+beta) * mean sq
    float rc = accs[1];
    out[0] = vq; out[1] = rc; out[2] = vq + rc;
  }
}

// ---------------------------------------------------------------------------

extern "C" void kernel_launch(void* const* d_in, const int* in_sizes, int n_in,
                              void* d_out, int out_size, void* d_ws, size_t ws_size,
                              hipStream_t stream)
{
  (void)in_sizes; (void)n_in; (void)out_size; (void)ws_size;

  // --- assumed input order: depth-first insertion order of setup_inputs() ---
  // 0: waveform; enc.stem(w,b)=1,2; enc.down[i](w,b)=3..12;
  // enc.blocks[i]=(w3,b3,w1,b1)=13..28; enc.proj(W,b)=29,30; codebooks=31..34;
  // dec.proj(W,b)=35,36; dec.up[i](w,b)=37..44; dec.blocks=45..60; dec.out(w,b)=61,62
  const float* wav = (const float*)d_in[0];
  auto F32 = [&](int i) { return (const float*)d_in[i]; };

  float* outF = (float*)d_out;        // [recon 320000][zq 204800][codes 12800][3 scalars]
  float* outZq    = outF + 320000;
  float* outCodes = outF + 524800;
  float* outLoss  = outF + 537600;

  // ---------------- workspace carve-up ----------------
  char* ws = (char*)d_ws;
  size_t off = 0;
  auto alloc = [&](size_t bytes) -> char* {
    char* p = ws + off; off = (off + bytes + 255) & ~(size_t)255; return p;
  };
  float* bufA = (float*)alloc(4ul * 128 * 80000 * 4);
  float* bufB = (float*)alloc(4ul * 128 * 80000 * 4);
  auto allocH = [&](size_t halves) { return (_Float16*)alloc(halves * 2); };

  _Float16* Wstem = allocH(128ul * 32);
  _Float16* Wd[5]; int KpD[5] = {512, 512, 512, 640, 640};
  for (int i = 0; i < 5; ++i) Wd[i] = allocH(128ul * KpD[i]);
  _Float16 *Web3[4], *Web1[4], *Wdb3[4], *Wdb1[4];
  for (int i = 0; i < 4; ++i) { Web3[i] = allocH(128ul * 384); Web1[i] = allocH(128ul * 128); }
  for (int i = 0; i < 4; ++i) { Wdb3[i] = allocH(128ul * 384); Wdb1[i] = allocH(128ul * 128); }
  _Float16* Wpe = allocH(64ul * 128);
  _Float16* Wpd = allocH(128ul * 64);
  _Float16* Wu[4]; int KpU[4] = {640, 640, 512, 512};
  for (int i = 0; i < 4; ++i) Wu[i] = allocH(128ul * KpU[i]);
  _Float16* Wout = allocH(16ul * 896);
  _Float16* E16[4]; for (int i = 0; i < 4; ++i) E16[i] = allocH(256ul * 64);

  float* enorm = (float*)alloc(4ul * 256 * 4);
  float* zT    = (float*)alloc(4ul * 64 * 800 * 4);
  float* rbuf  = (float*)alloc(4ul * 64 * 800 * 4);
  float* zqT   = (float*)alloc(4ul * 64 * 800 * 4);
  float* dotb  = (float*)alloc(4ul * 256 * 800 * 4);
  _Float16* Wc16 = allocH(1040ul * 2048);
  _Float16* Ws16 = allocH(1040ul * 2048);
  float* FrT = (float*)alloc(1310720ul * 4);
  float* Cxr = (float*)alloc(700000ul * 4);
  float* Cxi = (float*)alloc(700000ul * 4);
  float* Chr = (float*)alloc(700000ul * 4);
  float* Chi = (float*)alloc(700000ul * 4);
  float* accs = (float*)alloc(2 * 4);

  auto launch1 = [&](int n) { return dim3((unsigned)((n + 255) / 256)); };

  auto pack = [&](const float* w, _Float16* A, int Ma, int CoutAct, int Cin, int ksz,
                  int Kp, int flip) {
    int n = Ma * Kp;
    k_pack_w<<<launch1(n), 256, 0, stream>>>(w, A, CoutAct, Cin, ksz, Kp, flip, n);
  };
  ConvDesc P{}; P.mode = 0;
  auto CD = [](int Cin, int Lin, int ksz, int stride, int padL, int dil, int lhsDil,
               int elu) {
    ConvDesc c; c.mode = 1; c.Cin = Cin; c.Lin = Lin; c.ksz = ksz; c.stride = stride;
    c.padL = padL; c.dil = dil; c.lhsDil = lhsDil; c.elu = elu; return c;
  };
  auto gemm = [&](const _Float16* A, const float* B, const float* bias, float* C,
                  int Ma, int Mact, int N, int K, long sb, long sc, int batch,
                  ConvDesc cd, int act, int accum) {
    dim3 g((unsigned)((N + TNB - 1) / TNB), (unsigned)((Ma + TM - 1) / TM), (unsigned)batch);
    k_gemm<<<g, 256, 0, stream>>>(A, B, bias, C, Ma, Mact, N, K, sb, sc, cd, act, accum);
  };

  // ---------------- weight packing (f16) ----------------
  pack(F32(1), Wstem, 128, 128, 1, 7, 32, 0);
  int dK[5] = {4, 4, 4, 5, 5};
  for (int i = 0; i < 5; ++i) pack(F32(3 + 2 * i), Wd[i], 128, 128, 128, dK[i], KpD[i], 0);
  for (int i = 0; i < 4; ++i) {
    pack(F32(13 + 4 * i), Web3[i], 128, 128, 128, 3, 384, 0);
    pack(F32(15 + 4 * i), Web1[i], 128, 128, 128, 1, 128, 0);
    pack(F32(45 + 4 * i), Wdb3[i], 128, 128, 128, 3, 384, 0);
    pack(F32(47 + 4 * i), Wdb1[i], 128, 128, 128, 1, 128, 0);
  }
  pack(F32(29), Wpe, 64, 64, 128, 1, 128, 0);
  pack(F32(35), Wpd, 128, 128, 64, 1, 64, 0);
  int uK[4] = {5, 5, 4, 4};
  for (int i = 0; i < 4; ++i) pack(F32(37 + 2 * i), Wu[i], 128, 128, 128, uK[i], KpU[i], 1);
  pack(F32(61), Wout, 16, 1, 128, 7, 896, 0);
  for (int i = 0; i < 4; ++i) {
    pack(F32(31 + i), E16[i], 256, 256, 64, 1, 64, 0);
    k_enorm<<<1, 256, 0, stream>>>(F32(31 + i), enorm + i * 256, 256);
  }

  // ---------------- encoder ----------------
  gemm(Wstem, wav, F32(2), bufA, 128, 128, 80000, 32, 80000, 128l * 80000, 4,
       CD(1, 80000, 7, 1, 6, 1, 1, 0), 0, 0);
  gemm(Wd[0], bufA, F32(4), bufB, 128, 128, 80000, 512, 128l * 80000, 128l * 80000, 4,
       CD(128, 80000, 4, 1, 3, 1, 1, 1), 0, 0);
  gemm(Wd[1], bufB, F32(6), bufA, 128, 128, 40000, 512, 128l * 80000, 128l * 40000, 4,
       CD(128, 80000, 4, 2, 1, 1, 1, 1), 0, 0);
  gemm(Wd[2], bufA, F32(8), bufB, 128, 128, 20000, 512, 128l * 40000, 128l * 20000, 4,
       CD(128, 40000, 4, 2, 1, 1, 1, 1), 0, 0);
  gemm(Wd[3], bufB, F32(10), bufA, 128, 128, 4000, 640, 128l * 20000, 128l * 4000, 4,
       CD(128, 20000, 5, 5, 2, 1, 1, 1), 0, 0);
  gemm(Wd[4], bufA, F32(12), bufB, 128, 128, 800, 640, 128l * 4000, 128l * 800, 4,
       CD(128, 4000, 5, 5, 2, 1, 1, 1), 0, 0);
  for (int i = 0; i < 4; ++i) {                  // residual blocks (x += conv1(elu(conv3(elu x))))
    int dil = 1 << i;
    gemm(Web3[i], bufB, F32(14 + 4 * i), bufA, 128, 128, 800, 384, 128l * 800, 128l * 800,
         4, CD(128, 800, 3, 1, 2 * dil, dil, 1, 1), 0, 0);
    gemm(Web1[i], bufA, F32(16 + 4 * i), bufB, 128, 128, 800, 128, 128l * 800, 128l * 800,
         4, CD(128, 800, 1, 1, 0, 1, 1, 1), 0, 1);
  }
  gemm(Wpe, bufB, F32(30), zT, 64, 64, 800, 128, 128l * 800, 64l * 800, 4, P, 0, 0);

  // ---------------- RVQ ----------------
  (void)hipMemcpyAsync(rbuf, zT, 4ul * 64 * 800 * 4, hipMemcpyDeviceToDevice, stream);
  (void)hipMemsetAsync(zqT, 0, 4ul * 64 * 800 * 4, stream);
  (void)hipMemsetAsync(accs, 0, 2 * sizeof(float), stream);
  for (int s = 0; s < 4; ++s) {
    gemm(E16[s], rbuf, nullptr, dotb, 256, 256, 800, 64, 64l * 800, 256l * 800, 4, P, 0, 0);
    k_vq<<<launch1(4 * 800), 256, 0, stream>>>(dotb, enorm + s * 256, F32(31 + s),
                                               rbuf, zqT, outCodes, accs, s);
  }
  k_zq_out<<<launch1(4 * 800 * 64), 256, 0, stream>>>(zqT, outZq);

  // ---------------- decoder ----------------
  gemm(Wpd, zqT, F32(36), bufA, 128, 128, 800, 64, 64l * 800, 128l * 800, 4, P, 0, 0);
  gemm(Wu[0], bufA, F32(38), bufB, 128, 128, 4000, 640, 128l * 800, 128l * 4000, 4,
       CD(128, 800, 5, 1, 2, 1, 5, 1), 0, 0);
  gemm(Wu[1], bufB, F32(40), bufA, 128, 128, 20000, 640, 128l * 4000, 128l * 20000, 4,
       CD(128, 4000, 5, 1, 2, 1, 5, 1), 0, 0);
  gemm(Wu[2], bufA, F32(42), bufB, 128, 128, 40000, 512, 128l * 20000, 128l * 40000, 4,
       CD(128, 20000, 4, 1, 2, 1, 2, 1), 0, 0);
  gemm(Wu[3], bufB, F32(44), bufA, 128, 128, 80000, 512, 128l * 40000, 128l * 80000, 4,
       CD(128, 40000, 4, 1, 2, 1, 2, 1), 0, 0);
  for (int i = 0; i < 4; ++i) {
    int dil = 1 << i;
    gemm(Wdb3[i], bufA, F32(46 + 4 * i), bufB, 128, 128, 80000, 384, 128l * 80000,
         128l * 80000, 4, CD(128, 80000, 3, 1, 2 * dil, dil, 1, 1), 0, 0);
    gemm(Wdb1[i], bufB, F32(48 + 4 * i), bufA, 128, 128, 80000, 128, 128l * 80000,
         128l * 80000, 4, CD(128, 80000, 1, 1, 0, 1, 1, 1), 0, 1);
  }
  // final conv -> tanh -> recon written straight into d_out
  gemm(Wout, bufA, F32(62), outF, 16, 1, 80000, 896, 128l * 80000, 80000, 4,
       CD(128, 80000, 7, 1, 3, 1, 1, 0), 1, 0);

  // ---------------- multi-resolution STFT loss ----------------
  for (int si = 0; si < 3; ++si) {
    int n_fft = 512 << si, hop = n_fft / 4;
    int F = 80000 / hop + 1, Ncols = 4 * F;
    int bins = n_fft / 2 + 1, Mpad = ((bins + 15) / 16) * 16;
    k_dftw<<<launch1(Mpad * n_fft), 256, 0, stream>>>(Wc16, Ws16, n_fft, bins,
                                                      Mpad * n_fft);
    k_frames<<<launch1(n_fft * Ncols), 256, 0, stream>>>(wav, FrT, n_fft, hop, F, Ncols,
                                                         n_fft * Ncols);
    gemm(Wc16, FrT, nullptr, Cxr, Mpad, bins, Ncols, n_fft, 0, 0, 1, P, 0, 0);
    gemm(Ws16, FrT, nullptr, Cxi, Mpad, bins, Ncols, n_fft, 0, 0, 1, P, 0, 0);
    k_frames<<<launch1(n_fft * Ncols), 256, 0, stream>>>(outF, FrT, n_fft, hop, F, Ncols,
                                                         n_fft * Ncols);
    gemm(Wc16, FrT, nullptr, Chr, Mpad, bins, Ncols, n_fft, 0, 0, 1, P, 0, 0);
    gemm(Ws16, FrT, nullptr, Chi, Mpad, bins, Ncols, n_fft, 0, 0, 1, P, 0, 0);
    float invCount = 1.f / ((float)bins * (float)F * 4.f);
    k_sloss<<<launch1(bins * Ncols), 256, 0, stream>>>(Cxr, Cxi, Chr, Chi, accs + 1,
                                                       invCount, bins * Ncols);
  }
  k_final<<<1, 32, 0, stream>>>(accs, outLoss);
}
